// FlaxMixtralAttention_13151189860806
// MI455X (gfx1250) — compile-verified, matrix operand order checked by reading, not run on hardware
//
#include <hip/hip_runtime.h>

// ---------------------------------------------------------------------------
// Mixtral attention layer for MI455X (gfx1250, wave32, WMMA bf16 16x16x32)
//   B=2, S=2048, E=4096, H=32, KVH=8, D=128, GROUPS=4, fully causal
// ---------------------------------------------------------------------------

#define B_   2
#define S_   2048
#define E_   4096
#define H_   32
#define KVH_ 8
#define D_   128

typedef __attribute__((ext_vector_type(16))) __bf16 v16bf;
typedef __attribute__((ext_vector_type(8)))  float  v8f;

struct alignas(16) U4 { unsigned int x, y, z, w; };

union Frag16 {
    v16bf        v;
    unsigned int u[8];
    U4           q[2];
};

__device__ __forceinline__ unsigned short f2bf(float f) {
    unsigned int u = __float_as_uint(f);
    return (unsigned short)((u + 0x7FFFu + ((u >> 16) & 1u)) >> 16);
}

__device__ __forceinline__ unsigned int pack_bf2(float lo, float hi) {
    return ((unsigned int)f2bf(hi) << 16) | (unsigned int)f2bf(lo);
}

__device__ __forceinline__ v8f wmma_bf16(const Frag16& a, const Frag16& b, v8f c) {
    // D = A(16x32) * B(32x16) + C, fp32 accumulate
    return __builtin_amdgcn_wmma_f32_16x16x32_bf16(
        false, a.v, false, b.v, (short)0, c, false, false);
}

// ---------------------------------------------------------------------------
// Generic GEMM: C[M,N] = A[M,K] * B[K,N]; A is fp32 or raw-bf16, B fp32,
// C fp32. Block = 128x128 tile, 256 threads (8 waves, 2x4 wave grid),
// k-step = 32. LDS double-buffered with register-staged prefetch:
// one barrier per k-step, next tile's global loads overlap current WMMAs.
// ---------------------------------------------------------------------------
template <typename AT>
__global__ __launch_bounds__(256) void gemm_bf16_kernel(
        const AT* __restrict__ A, const float* __restrict__ Bw,
        float* __restrict__ C, int M, int N, int K)
{
    __shared__ unsigned int As [2][128][20];  // 128 rows x 32 bf16 (16 uints) + pad
    __shared__ unsigned int Bts[2][128][20];  // B transposed: 128 n-rows x 32 k

    const int t    = threadIdx.x;
    const int lane = t & 31;
    const int w    = t >> 5;
    const int wm   = w >> 2;       // 0..1 -> 64 rows each
    const int wn   = w & 3;        // 0..3 -> 32 cols each
    const int half = lane >> 4;
    const int l15  = lane & 15;
    const int mbase = blockIdx.y * 128;
    const int nbase = blockIdx.x * 128;

    // staging roles
    const int arow = t >> 1;          // 0..127
    const int aseg = t & 1;           // 0..1 (16 elements each)
    const int bk   = t >> 3;          // 0..31
    const int bn0  = (t & 7) * 16;    // 0..112

    // register staging for the in-flight tile
    float af[16];
    U4    au[2];
    float bfv[16];

    auto load_tile = [&](int ko) {
        if constexpr (sizeof(AT) == 4) {
            const float* ap = (const float*)A + (size_t)(mbase + arow) * K + ko + aseg * 16;
            ((float4*)af)[0] = ((const float4*)ap)[0];
            ((float4*)af)[1] = ((const float4*)ap)[1];
            ((float4*)af)[2] = ((const float4*)ap)[2];
            ((float4*)af)[3] = ((const float4*)ap)[3];
        } else {
            const unsigned short* ap =
                (const unsigned short*)A + (size_t)(mbase + arow) * K + ko + aseg * 16;
            au[0] = ((const U4*)ap)[0];
            au[1] = ((const U4*)ap)[1];
        }
        const float* bp = Bw + (size_t)(ko + bk) * N + nbase + bn0;
        ((float4*)bfv)[0] = ((const float4*)bp)[0];
        ((float4*)bfv)[1] = ((const float4*)bp)[1];
        ((float4*)bfv)[2] = ((const float4*)bp)[2];
        ((float4*)bfv)[3] = ((const float4*)bp)[3];
    };

    auto store_tile = [&](int buf) {
        unsigned int* dst = &As[buf][arow][aseg * 8];
        if constexpr (sizeof(AT) == 4) {
            dst[0] = pack_bf2(af[0],  af[1]);  dst[1] = pack_bf2(af[2],  af[3]);
            dst[2] = pack_bf2(af[4],  af[5]);  dst[3] = pack_bf2(af[6],  af[7]);
            dst[4] = pack_bf2(af[8],  af[9]);  dst[5] = pack_bf2(af[10], af[11]);
            dst[6] = pack_bf2(af[12], af[13]); dst[7] = pack_bf2(af[14], af[15]);
        } else {
            dst[0] = au[0].x; dst[1] = au[0].y; dst[2] = au[0].z; dst[3] = au[0].w;
            dst[4] = au[1].x; dst[5] = au[1].y; dst[6] = au[1].z; dst[7] = au[1].w;
        }
        unsigned short* bt = (unsigned short*)&Bts[buf][0][0];
        #pragma unroll
        for (int i = 0; i < 16; i++)
            bt[(size_t)(bn0 + i) * 40 + bk] = f2bf(bfv[i]);
    };

    v8f acc[4][2] = {};

    load_tile(0);
    store_tile(0);
    __syncthreads();

    const int nk = K >> 5;
    for (int i = 0; i < nk; i++) {
        const int cur = i & 1;
        if (i + 1 < nk) load_tile((i + 1) << 5);   // prefetch next tile (VMEM in flight)

        // ---- fragments + WMMA from buffer `cur` ----
        Frag16 a[4], b[2];
        #pragma unroll
        for (int ms = 0; ms < 4; ms++) {
            const int row = wm * 64 + ms * 16 + l15;
            a[ms].q[0] = *(const U4*)&As[cur][row][half * 4];
            a[ms].q[1] = *(const U4*)&As[cur][row][8 + half * 4];
        }
        #pragma unroll
        for (int ns = 0; ns < 2; ns++) {
            const int n = wn * 32 + ns * 16 + l15;
            b[ns].q[0] = *(const U4*)&Bts[cur][n][half * 4];
            b[ns].q[1] = *(const U4*)&Bts[cur][n][8 + half * 4];
        }
        #pragma unroll
        for (int ms = 0; ms < 4; ms++)
            #pragma unroll
            for (int ns = 0; ns < 2; ns++)
                acc[ms][ns] = wmma_bf16(a[ms], b[ns], acc[ms][ns]);

        if (i + 1 < nk) {
            store_tile(cur ^ 1);   // safe: other waves finished reading buf cur^1
            __syncthreads();       // before the barrier at end of iteration i-1
        }
    }

    // ---- epilogue: C layout = rows r (lanes 0-15) / r+8 (lanes 16-31) ----
    #pragma unroll
    for (int ms = 0; ms < 4; ms++)
        for (int ns = 0; ns < 2; ns++)
            #pragma unroll
            for (int r = 0; r < 8; r++) {
                const int crow = mbase + wm * 64 + ms * 16 + r + 8 * half;
                const int ccol = nbase + wn * 32 + ns * 16 + l15;
                C[(size_t)crow * N + ccol] = acc[ms][ns][r];
            }
}

// ---------------------------------------------------------------------------
// RoPE + transpose + bf16 convert.  Q: [B,S,H,D] fp32 -> [B,H,S,D] bf16
// ---------------------------------------------------------------------------
__global__ __launch_bounds__(256) void rope_q_kernel(
        const float* __restrict__ Q, unsigned short* __restrict__ Qb)
{
    const int idx = blockIdx.x * 256 + threadIdx.x;   // B*S*H*64 threads
    const int d2 = idx & 63;
    const int h  = (idx >> 6) & 31;
    const int s  = (idx >> 11) & 2047;
    const int b  = idx >> 22;
    const size_t in = ((size_t)(b * S_ + s) * H_ + h) * D_ + d2;
    const float q1 = Q[in], q2 = Q[in + 64];
    const float inv = __expf(-(float)d2 * (9.210340371976184f / 64.0f)); // theta^(-d2/64)
    const float f = (float)s * inv;
    float sn, cs; __sincosf(f, &sn, &cs);
    const size_t op = ((size_t)(b * H_ + h) * S_ + s) * D_ + d2;
    Qb[op]      = f2bf(q1 * cs - q2 * sn);
    Qb[op + 64] = f2bf(q2 * cs + q1 * sn);
}

// K: rope + transpose; V: transpose only.  [B,S,KVH,D] -> [B,KVH,S,D] bf16
__global__ __launch_bounds__(256) void rope_kv_kernel(
        const float* __restrict__ K, const float* __restrict__ V,
        unsigned short* __restrict__ Kb, unsigned short* __restrict__ Vb)
{
    const int idx = blockIdx.x * 256 + threadIdx.x;   // B*S*KVH*64 threads
    const int d2 = idx & 63;
    const int hk = (idx >> 6) & 7;
    const int s  = (idx >> 9) & 2047;
    const int b  = idx >> 20;
    const size_t in = ((size_t)(b * S_ + s) * KVH_ + hk) * D_ + d2;
    const float k1 = K[in], k2 = K[in + 64];
    const float inv = __expf(-(float)d2 * (9.210340371976184f / 64.0f));
    const float f = (float)s * inv;
    float sn, cs; __sincosf(f, &sn, &cs);
    const size_t op = ((size_t)(b * KVH_ + hk) * S_ + s) * D_ + d2;
    Kb[op]      = f2bf(k1 * cs - k2 * sn);
    Kb[op + 64] = f2bf(k2 * cs + k1 * sn);
    Vb[op]      = f2bf(V[in]);
    Vb[op + 64] = f2bf(V[in + 64]);
}

// ---------------------------------------------------------------------------
// Flash attention (causal, GQA). Block = 64 queries of one (b,h), 128 thr
// (4 waves, 16 q-rows per wave). Key tile = 32. All matmuls bf16 WMMA.
// ---------------------------------------------------------------------------
__global__ __launch_bounds__(128) void flash_attn_kernel(
        const unsigned short* __restrict__ Qb,
        const unsigned short* __restrict__ Kb,
        const unsigned short* __restrict__ Vb,
        unsigned short* __restrict__ Ob)
{
    __shared__ unsigned int Kl[32][68];      // 32 keys x 128 d (bf16), padded
    __shared__ unsigned int Vt[128][20];     // V transposed: 128 d x 32 keys
    __shared__ unsigned int Pl[4][16][16];   // per-wave P: 16 q x 32 keys

    const int t    = threadIdx.x;
    const int lane = t & 31;
    const int w    = t >> 5;
    const int half = lane >> 4;
    const int l15  = lane & 15;
    const int qt = blockIdx.x, h = blockIdx.y, b = blockIdx.z;
    const int hk = h >> 2;                               // GQA: 4 q-heads per kv-head
    const float scale = 0.08838834764831845f;            // 1/sqrt(128)

    // Q fragments straight from global in A-fragment layout
    const unsigned int* qp =
        (const unsigned int*)(Qb + ((size_t)(b * H_ + h) * S_) * D_);
    const int qrow_a = qt * 64 + w * 16 + l15;
    Frag16 qf[4];
    #pragma unroll
    for (int ks = 0; ks < 4; ks++) {
        qf[ks].q[0] = *(const U4*)(qp + (size_t)qrow_a * 64 + ks * 16 + half * 4);
        qf[ks].q[1] = *(const U4*)(qp + (size_t)qrow_a * 64 + ks * 16 + 8 + half * 4);
    }

    v8f o[8] = {};
    float m_i[8], l_i[8];
    #pragma unroll
    for (int r = 0; r < 8; r++) { m_i[r] = -3.0e38f; l_i[r] = 0.0f; }

    const unsigned short* kbase = Kb + (size_t)(b * KVH_ + hk) * S_ * D_;
    const unsigned short* vbase = Vb + (size_t)(b * KVH_ + hk) * S_ * D_;
    const int nkt = (qt + 1) * 2;   // causal: keys up to (qt+1)*64

    for (int kt = 0; kt < nkt; kt++) {
        __syncthreads();
        // ---- stage K tile (rows as-is) and V tile (transposed) ----
        {
            const int key = t >> 2;
            const int s0  = (t & 3) * 16;   // uint column
            const unsigned int* krow =
                (const unsigned int*)(kbase + (size_t)(kt * 32 + key) * D_);
            U4 k0 = ((const U4*)(krow + s0))[0];
            U4 k1 = ((const U4*)(krow + s0))[1];
            U4 k2 = ((const U4*)(krow + s0))[2];
            U4 k3 = ((const U4*)(krow + s0))[3];
            unsigned int* kd = &Kl[key][s0];
            kd[0]=k0.x; kd[1]=k0.y; kd[2]=k0.z; kd[3]=k0.w;
            kd[4]=k1.x; kd[5]=k1.y; kd[6]=k1.z; kd[7]=k1.w;
            kd[8]=k2.x; kd[9]=k2.y; kd[10]=k2.z; kd[11]=k2.w;
            kd[12]=k3.x; kd[13]=k3.y; kd[14]=k3.z; kd[15]=k3.w;

            const unsigned int* vrow =
                (const unsigned int*)(vbase + (size_t)(kt * 32 + key) * D_);
            U4 v0 = ((const U4*)(vrow + s0))[0];
            U4 v1 = ((const U4*)(vrow + s0))[1];
            U4 v2 = ((const U4*)(vrow + s0))[2];
            U4 v3 = ((const U4*)(vrow + s0))[3];
            unsigned int vv[16] = { v0.x,v0.y,v0.z,v0.w, v1.x,v1.y,v1.z,v1.w,
                                    v2.x,v2.y,v2.z,v2.w, v3.x,v3.y,v3.z,v3.w };
            unsigned short* vt = (unsigned short*)&Vt[0][0];
            #pragma unroll
            for (int i = 0; i < 16; i++) {
                const int c = s0 + i;
                vt[(size_t)(2 * c)     * 40 + key] = (unsigned short)(vv[i] & 0xFFFFu);
                vt[(size_t)(2 * c + 1) * 40 + key] = (unsigned short)(vv[i] >> 16);
            }
        }
        __syncthreads();

        // ---- S = Q * K^T (16x32 per wave) ----
        v8f s0acc = {}, s1acc = {};
        #pragma unroll
        for (int ks = 0; ks < 4; ks++) {
            Frag16 b0, b1;
            b0.q[0] = *(const U4*)&Kl[l15][ks * 16 + half * 4];
            b0.q[1] = *(const U4*)&Kl[l15][ks * 16 + 8 + half * 4];
            b1.q[0] = *(const U4*)&Kl[16 + l15][ks * 16 + half * 4];
            b1.q[1] = *(const U4*)&Kl[16 + l15][ks * 16 + 8 + half * 4];
            s0acc = wmma_bf16(qf[ks], b0, s0acc);
            s1acc = wmma_bf16(qf[ks], b1, s1acc);
        }

        // ---- online softmax (rows split across lane halves per C layout) ----
        const int ktb = kt * 32;
        float alpha[8];
        #pragma unroll
        for (int r = 0; r < 8; r++) {
            const int qr = qt * 64 + w * 16 + r + 8 * half;
            float sv0 = s0acc[r] * scale;
            float sv1 = s1acc[r] * scale;
            if (ktb + l15 > qr)      sv0 = -3.0e38f;
            if (ktb + 16 + l15 > qr) sv1 = -3.0e38f;
            float mr = fmaxf(sv0, sv1);
            #pragma unroll
            for (int off = 1; off < 16; off <<= 1)
                mr = fmaxf(mr, __shfl_xor(mr, off, 32));
            const float mn = fmaxf(m_i[r], mr);
            const float al = __expf(m_i[r] - mn);
            const float p0 = __expf(sv0 - mn);
            const float p1 = __expf(sv1 - mn);
            float rs = p0 + p1;
            #pragma unroll
            for (int off = 1; off < 16; off <<= 1)
                rs += __shfl_xor(rs, off, 32);
            l_i[r] = l_i[r] * al + rs;
            m_i[r] = mn;
            alpha[r] = al;
            unsigned short* pp = (unsigned short*)&Pl[w][r + 8 * half][0];
            pp[l15]      = f2bf(p0);
            pp[16 + l15] = f2bf(p1);
        }
        #pragma unroll
        for (int j = 0; j < 8; j++)
            #pragma unroll
            for (int r = 0; r < 8; r++)
                o[j][r] *= alpha[r];

        // ---- O += P * V (P via LDS round-trip into A-fragment layout) ----
        Frag16 pa;
        pa.q[0] = *(const U4*)&Pl[w][l15][half * 4];
        pa.q[1] = *(const U4*)&Pl[w][l15][8 + half * 4];
        #pragma unroll
        for (int j = 0; j < 8; j++) {
            Frag16 vb;
            const int d = j * 16 + l15;
            vb.q[0] = *(const U4*)&Vt[d][half * 4];
            vb.q[1] = *(const U4*)&Vt[d][8 + half * 4];
            o[j] = wmma_bf16(pa, vb, o[j]);
        }
    }

    // ---- epilogue: normalize and emit bf16 [B*S, H*D] for the O-proj GEMM ----
    #pragma unroll
    for (int r = 0; r < 8; r++) {
        const float inv = 1.0f / l_i[r];
        const int qr = qt * 64 + w * 16 + r + 8 * half;
        const size_t rowo = (size_t)(b * S_ + qr) * (H_ * D_) + (size_t)h * D_;
        #pragma unroll
        for (int j = 0; j < 8; j++)
            Ob[rowo + j * 16 + l15] = f2bf(o[j][r] * inv);
    }
}

// ---------------------------------------------------------------------------
// Launch
// ---------------------------------------------------------------------------
extern "C" void kernel_launch(void* const* d_in, const int* in_sizes, int n_in,
                              void* d_out, int out_size, void* d_ws, size_t ws_size,
                              hipStream_t stream)
{
    const float* hidden = (const float*)d_in[0];
    const float* Wq     = (const float*)d_in[1];
    const float* Wk     = (const float*)d_in[2];
    const float* Wv     = (const float*)d_in[3];
    const float* Wo     = (const float*)d_in[4];
    float*       out    = (float*)d_out;

    char* ws = (char*)d_ws;
    size_t off = 0;
    float* Qbuf = (float*)(ws + off);              off += (size_t)4096 * 4096 * 4;
    float* Kbuf = (float*)(ws + off);              off += (size_t)4096 * 1024 * 4;
    float* Vbuf = (float*)(ws + off);              off += (size_t)4096 * 1024 * 4;
    unsigned short* Qb = (unsigned short*)(ws + off); off += (size_t)4096 * 4096 * 2;
    unsigned short* Kb = (unsigned short*)(ws + off); off += (size_t)4096 * 1024 * 2;
    unsigned short* Vb = (unsigned short*)(ws + off); off += (size_t)4096 * 1024 * 2;
    unsigned short* Ab = (unsigned short*)(ws + off); off += (size_t)4096 * 4096 * 2;
    (void)ws_size; (void)in_sizes; (void)n_in; (void)out_size;

    // QKV projections (fp32 -> bf16 WMMA -> fp32)
    gemm_bf16_kernel<float><<<dim3(32, 32), 256, 0, stream>>>(hidden, Wq, Qbuf, 4096, 4096, 4096);
    gemm_bf16_kernel<float><<<dim3( 8, 32), 256, 0, stream>>>(hidden, Wk, Kbuf, 4096, 1024, 4096);
    gemm_bf16_kernel<float><<<dim3( 8, 32), 256, 0, stream>>>(hidden, Wv, Vbuf, 4096, 1024, 4096);

    // RoPE + transpose + bf16
    rope_q_kernel <<<32768, 256, 0, stream>>>(Qbuf, Qb);
    rope_kv_kernel<<< 8192, 256, 0, stream>>>(Kbuf, Vbuf, Kb, Vb);

    // Causal GQA flash attention
    flash_attn_kernel<<<dim3(32, 32, 2), 128, 0, stream>>>(Qb, Kb, Vb, Ab);

    // Output projection (bf16 A x fp32 W -> fp32 out)
    gemm_bf16_kernel<unsigned short><<<dim3(32, 32), 256, 0, stream>>>(Ab, Wo, out, 4096, 4096, 4096);
}
